// RNNEncoder_58360015618645
// MI455X (gfx1250) — matrix operand order repaired and checked by reading, not compiled, not run on hardware
//
#include <hip/hip_runtime.h>

typedef __attribute__((ext_vector_type(16))) _Float16 v16h;
typedef __attribute__((ext_vector_type(8)))  _Float16 v8h;
typedef __attribute__((ext_vector_type(8)))  float    v8f;

#define B_SZ   8192
#define T_SZ   26
#define DIN_SZ 2
#define H_SZ   512
#define N3H    1536
#define MTILE  32
#define KT_STEPS 16          // 512 / 32
#define HPAD   520           // 512 + 8 halves pad -> 1040B row stride (bank-spread)
#define THREADS 512          // 16 waves

// ---------------------------------------------------------------------------
// Prep 1: A2 = W_emb @ K  (2 x 1536), bx = b_emb @ K + b_in  (1536)
// ---------------------------------------------------------------------------
__global__ void prep_bias_kernel(const float* __restrict__ W_emb,
                                 const float* __restrict__ b_emb,
                                 const float* __restrict__ K,
                                 const float* __restrict__ b_in,
                                 float* __restrict__ A2_0,
                                 float* __restrict__ A2_1,
                                 float* __restrict__ bx) {
    int n = blockIdx.x * blockDim.x + threadIdx.x;
    if (n >= N3H) return;
    float a0 = 0.f, a1 = 0.f, bb = 0.f;
    for (int k = 0; k < H_SZ; ++k) {
        float kv = K[k * N3H + n];
        a0 += W_emb[k] * kv;            // W_emb[0][k]
        a1 += W_emb[H_SZ + k] * kv;     // W_emb[1][k]
        bb += b_emb[k] * kv;
    }
    A2_0[n] = a0;
    A2_1[n] = a1;
    bx[n]   = bb + b_in[n];
}

// ---------------------------------------------------------------------------
// Prep 2: pack RK (f32 [512,1536]) into f16 WMMA B fragments.
// Fragment f = nt*16 + kt holds the 32x16 tile (K rows kt*32.., N cols nt*16..)
// laid out per lane: n = lane&15, K = e + (lane>=16 ? 16 : 0)  (e = 0..15).
// ---------------------------------------------------------------------------
__global__ void pack_rk_kernel(const float* __restrict__ RK,
                               _Float16* __restrict__ Bpack) {
    int tid = blockIdx.x * blockDim.x + threadIdx.x;
    if (tid >= H_SZ * N3H) return;
    int f    = tid >> 9;          // fragment index
    int r    = tid & 511;
    int lane = r >> 4;
    int e    = r & 15;
    int nt   = f >> 4;
    int kt   = f & 15;
    int k = kt * 32 + e + ((lane & 16) ? 16 : 0);
    int n = nt * 16 + (lane & 15);
    Bpack[tid] = (_Float16)RK[k * N3H + n];
}

// ---------------------------------------------------------------------------
// Main fused GRU: each block owns 32 batch rows for all 26 timesteps.
// Each wave owns matching z/r/h column slices so gates are wave-local.
// ---------------------------------------------------------------------------
__global__ __launch_bounds__(THREADS)
void gru_main_kernel(const float* __restrict__ x,          // [B, T, 2]
                     const _Float16* __restrict__ Bpack,   // packed RK f16
                     const float* __restrict__ A2_0,
                     const float* __restrict__ A2_1,
                     const float* __restrict__ bx,
                     const float* __restrict__ b_rec,
                     float* __restrict__ out) {            // [B, T, 512]
    __shared__ _Float16 hbuf[MTILE * HPAD] __attribute__((aligned(16)));
    __shared__ float    xs[MTILE * T_SZ * DIN_SZ];

    const int tid  = threadIdx.x;
    const int wave = tid >> 5;
    const int lane = tid & 31;
    const int hi16 = (lane >> 4) & 1;
    const int ncol = lane & 15;
    const int row0 = blockIdx.x * MTILE;

    // zero hidden state
    for (int i = tid; i < MTILE * HPAD; i += THREADS) hbuf[i] = (_Float16)0.f;
    // stage x tile (contiguous copy: xs[m][t][d] matches global order)
    const float* xg = x + (size_t)row0 * (T_SZ * DIN_SZ);
    for (int i = tid; i < MTILE * T_SZ * DIN_SZ; i += THREADS) xs[i] = xg[i];

    // per-lane constants for the 6 owned column tiles (z/r/h x 2)
    float a0c[3][2], a1c[3][2], bxc[3][2], brc[3][2];
#pragma unroll
    for (int g = 0; g < 3; ++g)
#pragma unroll
        for (int p = 0; p < 2; ++p) {
            int n = g * H_SZ + (2 * wave + p) * 16 + ncol;
            a0c[g][p] = A2_0[n];
            a1c[g][p] = A2_1[n];
            bxc[g][p] = bx[n];
            brc[g][p] = b_rec[n];
        }

    __syncthreads();

    // A-fragment LDS sub-chunk offsets (contiguous K, matches Bpack):
    // lanes 0-15: K {0..15}; lanes 16-31: K {16..31}
    const int c0 = hi16 ? 16 : 0;
    const int c1 = c0 + 8;

    for (int t = 0; t < T_SZ; ++t) {
        v8f acc[2][6];
#pragma unroll
        for (int mt = 0; mt < 2; ++mt)
#pragma unroll
            for (int s = 0; s < 6; ++s) { v8f z = {}; acc[mt][s] = z; }

#pragma unroll 4
        for (int kt = 0; kt < KT_STEPS; ++kt) {
            v16h afr[2];
#pragma unroll
            for (int mt = 0; mt < 2; ++mt) {
                const _Float16* hp = &hbuf[(mt * 16 + ncol) * HPAD + kt * 32];
                v8h lo = *(const v8h*)(hp + c0);
                v8h hi = *(const v8h*)(hp + c1);
                afr[mt] = __builtin_shufflevector(lo, hi,
                          0,1,2,3,4,5,6,7,8,9,10,11,12,13,14,15);
            }
#pragma unroll
            for (int s = 0; s < 6; ++s) {
                const int g = s >> 1, p = s & 1;
                const int ntg = g * 32 + 2 * wave + p;   // global n-tile
                const v16h bfr = *(const v16h*)(Bpack +
                    ((size_t)(ntg * KT_STEPS + kt) << 9) + (lane << 4));
#pragma unroll
                for (int mt = 0; mt < 2; ++mt)
                    acc[mt][s] = __builtin_amdgcn_wmma_f32_16x16x32_f16(
                        false, afr[mt], false, bfr, (short)0, acc[mt][s],
                        false, false);
            }
        }

        __syncthreads();   // all matmul reads of h done before state update

        // gates + state update on owned columns (all in registers)
#pragma unroll
        for (int mt = 0; mt < 2; ++mt)
#pragma unroll
            for (int p = 0; p < 2; ++p) {
                const int ncs = (2 * wave + p) * 16 + ncol;  // state col [0,512)
#pragma unroll
                for (int v = 0; v < 8; ++v) {
                    const int m = mt * 16 + v + (hi16 ? 8 : 0);
                    const float x0 = xs[(m * T_SZ + t) * DIN_SZ + 0];
                    const float x1 = xs[(m * T_SZ + t) * DIN_SZ + 1];
                    const float mz = x0 * a0c[0][p] + x1 * a1c[0][p] + bxc[0][p];
                    const float mr = x0 * a0c[1][p] + x1 * a1c[1][p] + bxc[1][p];
                    const float mh = x0 * a0c[2][p] + x1 * a1c[2][p] + bxc[2][p];
                    const float iz = acc[mt][0 + p][v] + brc[0][p];
                    const float ir = acc[mt][2 + p][v] + brc[1][p];
                    const float ih = acc[mt][4 + p][v] + brc[2][p];
                    const float zg = 1.f / (1.f + __expf(-(mz + iz)));
                    const float rg = 1.f / (1.f + __expf(-(mr + ir)));
                    const float pre = mh + rg * ih;
                    const float e2 = __expf(-2.f * pre);
                    const float hh = (1.f - e2) / (1.f + e2);
                    const float hold = (float)hbuf[m * HPAD + ncs];
                    const float hnew = zg * hold + (1.f - zg) * hh;
                    hbuf[m * HPAD + ncs] = (_Float16)hnew;
                    out[((size_t)(row0 + m) * T_SZ + t) * H_SZ + ncs] = hnew;
                }
            }
        __syncthreads();   // state writes visible before next step's matmul
    }
}

// ---------------------------------------------------------------------------
extern "C" void kernel_launch(void* const* d_in, const int* in_sizes, int n_in,
                              void* d_out, int out_size, void* d_ws, size_t ws_size,
                              hipStream_t stream) {
    const float* x     = (const float*)d_in[0];
    const float* W_emb = (const float*)d_in[1];
    const float* b_emb = (const float*)d_in[2];
    const float* K     = (const float*)d_in[3];
    const float* RK    = (const float*)d_in[4];
    const float* b_in  = (const float*)d_in[5];
    const float* b_rec = (const float*)d_in[6];
    float* out = (float*)d_out;

    // workspace layout: [f16 packed RK | A2_0 | A2_1 | bx] (~1.54 MB)
    _Float16* Bpack = (_Float16*)d_ws;
    float* fs   = (float*)((char*)d_ws + (size_t)H_SZ * N3H * sizeof(_Float16));
    float* A2_0 = fs;
    float* A2_1 = fs + N3H;
    float* bxp  = fs + 2 * N3H;

    prep_bias_kernel<<<(N3H + 255) / 256, 256, 0, stream>>>(
        W_emb, b_emb, K, b_in, A2_0, A2_1, bxp);
    pack_rk_kernel<<<(H_SZ * N3H + 255) / 256, 256, 0, stream>>>(RK, Bpack);
    gru_main_kernel<<<B_SZ / MTILE, THREADS, 0, stream>>>(
        x, Bpack, A2_0, A2_1, bxp, b_rec, out);
}
